// GNN_graphmvp_49572512531053
// MI455X (gfx1250) — matrix operand order repaired and checked by reading, not compiled, not run on hardware
//
#include <hip/hip_runtime.h>
#include <hip/hip_bf16.h>

// ---------------------------------------------------------------------------
// GIN (graphmvp) forward for MI455X / gfx1250 (wave32, WMMA bf16 16x16x32).
// ---------------------------------------------------------------------------

typedef __attribute__((ext_vector_type(16))) __bf16 v16bf;
typedef __attribute__((ext_vector_type(8)))  __bf16 v8bf;
typedef __attribute__((ext_vector_type(8)))  float  v8f;

#define EMB 300
#define GRID1D(n) ((unsigned)((((long long)(n)) + 255) / 256))

// ---- atom encoder: h[n,d] = sum_f atom_emb[f, x_feat[n,f], d] --------------
__global__ __launch_bounds__(256) void atom_enc_kernel(
    const int* __restrict__ xf, const float* __restrict__ emb,
    float* __restrict__ h, long long total) {
  long long idx = (long long)blockIdx.x * blockDim.x + threadIdx.x;
  if (idx >= total) return;
  int n = (int)(idx / EMB);
  int d = (int)(idx % EMB);
  float s = 0.f;
#pragma unroll
  for (int f = 0; f < 9; ++f) {
    int v = xf[n * 9 + f];
    s += emb[((size_t)f * 119 + v) * EMB + d];
  }
  h[idx] = s;
}

// ---- z = (1 + eps[l]) * h --------------------------------------------------
__global__ __launch_bounds__(256) void zinit_kernel(
    const float* __restrict__ h, const float* __restrict__ eps, int l,
    float* __restrict__ z, long long total) {
  long long idx = (long long)blockIdx.x * blockDim.x + threadIdx.x;
  if (idx >= total) return;
  float s = 1.f + eps[l];
  z[idx] = s * h[idx];
}

// ---- message + scatter: z[dst] += relu(h[src] + bond_emb_sum) --------------
__global__ __launch_bounds__(256) void message_kernel(
    const int* __restrict__ ei, const int* __restrict__ ea,
    const float* __restrict__ bond, const float* __restrict__ h,
    float* __restrict__ z, int E, long long total) {
  long long idx = (long long)blockIdx.x * blockDim.x + threadIdx.x;
  if (idx >= total) return;
  int e = (int)(idx / EMB);
  int d = (int)(idx % EMB);
  int src = ei[e];
  int dst = ei[E + e];
  float s = 0.f;
#pragma unroll
  for (int t = 0; t < 3; ++t) {
    int a = ea[e * 3 + t];
    s += bond[((size_t)t * 6 + a) * EMB + d];
  }
  float m = h[(size_t)src * EMB + d] + s;
  m = m > 0.f ? m : 0.f;
  // lowers to global_atomic_add_f32 (no CAS loop)
  __hip_atomic_fetch_add(&z[(size_t)dst * EMB + d], m,
                         __ATOMIC_RELAXED, __HIP_MEMORY_SCOPE_AGENT);
}

// ---- f32 -> bf16 with K padding: dst[n,KP], zeros for k >= C ---------------
__global__ __launch_bounds__(256) void cvt_pad_kernel(
    const float* __restrict__ src, __bf16* __restrict__ dst,
    int C, int CP, long long total) {
  long long idx = (long long)blockIdx.x * blockDim.x + threadIdx.x;
  if (idx >= total) return;
  int n = (int)(idx / CP);
  int k = (int)(idx % CP);
  dst[idx] = (k < C) ? (__bf16)src[(size_t)n * C + k] : (__bf16)0.f;
}

// ---- weight prep: W[K,NOUT] f32 -> Wt[NP,KP] bf16 transposed, zero-pad -----
__global__ __launch_bounds__(256) void prep_w_kernel(
    const float* __restrict__ W, __bf16* __restrict__ Wt,
    int K, int NOUT, int KP, long long total) {
  long long idx = (long long)blockIdx.x * blockDim.x + threadIdx.x;
  if (idx >= total) return;
  int n = (int)(idx / KP);
  int k = (int)(idx % KP);
  Wt[idx] = (n < NOUT && k < K) ? (__bf16)W[(size_t)k * NOUT + n] : (__bf16)0.f;
}

// ---- WMMA GEMM: C[M,NOUT] = A[M,KP](bf16) x Bt[NP,KP](bf16)^T + bias -------
// One wave computes one 16x16 tile; K-loop in chunks of 32 (bf16 WMMA).
__global__ __launch_bounds__(256) void gemm_bf16_kernel(
    const __bf16* __restrict__ A, const __bf16* __restrict__ Bt,
    const float* __restrict__ bias, float* __restrict__ C,
    int M, int KP, int NOUT, int ntilesN) {
  int wave = blockIdx.x * (blockDim.x >> 5) + (threadIdx.x >> 5);
  int lane = threadIdx.x & 31;
  int tM = wave / ntilesN;
  int tN = wave % ntilesN;
  if (tM * 16 >= M) return;

  int half = lane >> 4;   // 0: lanes 0-15, 1: lanes 16-31
  int l16  = lane & 15;

  // A fragment: contiguous k order, not the ISA table order — lane row = l16,
  //   lanes 0-15 hold K=0..15, lanes 16-31 hold K=16..31 (contiguous, matches B)
  const __bf16* arow = A + (size_t)(tM * 16 + l16) * KP + half * 16;
  // B fragment: contiguous k order, same as A — lane col = l16,
  //   lanes 0-15 hold K=0..15, lanes 16-31 hold K=16..31 (contiguous in Wt)
  const __bf16* brow = Bt + (size_t)(tN * 16 + l16) * KP + half * 16;

  v8f acc = {};
  int nchunks = KP >> 5;
  for (int kc = 0; kc < nchunks; ++kc) {
    v8bf alo = *(const v8bf*)(arow);        // K +0..7   (16B load)
    v8bf ahi = *(const v8bf*)(arow + 8);    // K +8..15  (16B load)
    v16bf a = __builtin_shufflevector(alo, ahi,
        0, 1, 2, 3, 4, 5, 6, 7, 8, 9, 10, 11, 12, 13, 14, 15);
    v16bf b = *(const v16bf*)(brow);        // 16 consecutive K (32B load)
    acc = __builtin_amdgcn_wmma_f32_16x16x32_bf16(
        /*neg_a=*/false, a, /*neg_b=*/false, b,
        /*c_mod=*/(short)0, acc, /*reuse_a=*/false, /*reuse_b=*/false);
    arow += 32;
    brow += 32;
  }

  int col = tN * 16 + l16;
  if (col < NOUT) {
    float bs = bias[col];
    int row0 = tM * 16 + half * 8;  // C layout: VGPR r -> M = r + half*8
#pragma unroll
    for (int r = 0; r < 8; ++r) {
      int row = row0 + r;
      if (row < M) C[(size_t)row * NOUT + col] = acc[r] + bs;
    }
  }
}

// ---- zero fill -------------------------------------------------------------
__global__ __launch_bounds__(256) void zero_kernel(float* __restrict__ p, int n) {
  int i = blockIdx.x * blockDim.x + threadIdx.x;
  if (i < n) p[i] = 0.f;
}

// ---- column stats: sums[c] += sum(X[:,c]), sums[C+c] += sum(X[:,c]^2) ------
__global__ __launch_bounds__(256) void colstats_kernel(
    const float* __restrict__ X, float* __restrict__ sums,
    int nrows, int ncols, int rowsPerBlock) {
  int r0 = blockIdx.x * rowsPerBlock;
  int r1 = r0 + rowsPerBlock;
  if (r1 > nrows) r1 = nrows;
  for (int c = threadIdx.x; c < ncols; c += blockDim.x) {
    float s = 0.f, q = 0.f;
    for (int r = r0; r < r1; ++r) {
      float v = X[(size_t)r * ncols + c];  // coalesced across threads
      s += v;
      q += v * v;
    }
    __hip_atomic_fetch_add(&sums[c], s, __ATOMIC_RELAXED, __HIP_MEMORY_SCOPE_AGENT);
    __hip_atomic_fetch_add(&sums[ncols + c], q, __ATOMIC_RELAXED, __HIP_MEMORY_SCOPE_AGENT);
  }
}

// ---- BN + ReLU + convert to padded bf16 (mid-MLP activation) ---------------
__global__ __launch_bounds__(256) void bn_relu_cvt_kernel(
    const float* __restrict__ T, const float* __restrict__ sums,
    const float* __restrict__ g, const float* __restrict__ be,
    __bf16* __restrict__ U, int C, int CP, float invN, long long total) {
  long long idx = (long long)blockIdx.x * blockDim.x + threadIdx.x;
  if (idx >= total) return;
  int n = (int)(idx / CP);
  int c = (int)(idx % CP);
  __bf16 o = (__bf16)0.f;
  if (c < C) {
    float mu  = sums[c] * invN;
    float var = sums[C + c] * invN - mu * mu;
    float x = T[(size_t)n * C + c];
    float y = (x - mu) * __frsqrt_rn(var + 1e-5f) * g[c] + be[c];
    y = y > 0.f ? y : 0.f;
    o = (__bf16)y;
  }
  U[idx] = o;
}

// ---- outer BN (+optional ReLU) -> f32 h ------------------------------------
__global__ __launch_bounds__(256) void bn_final_kernel(
    const float* __restrict__ T, const float* __restrict__ sums,
    const float* __restrict__ g, const float* __restrict__ be,
    float* __restrict__ h, int C, float invN, int doRelu, long long total) {
  long long idx = (long long)blockIdx.x * blockDim.x + threadIdx.x;
  if (idx >= total) return;
  int c = (int)(idx % C);
  float mu  = sums[c] * invN;
  float var = sums[C + c] * invN - mu * mu;
  float y = (T[idx] - mu) * __frsqrt_rn(var + 1e-5f) * g[c] + be[c];
  if (doRelu) y = y > 0.f ? y : 0.f;
  h[idx] = y;
}

// ---------------------------------------------------------------------------
extern "C" void kernel_launch(void* const* d_in, const int* in_sizes, int n_in,
                              void* d_out, int out_size, void* d_ws, size_t ws_size,
                              hipStream_t stream) {
  (void)n_in; (void)out_size; (void)ws_size;

  const int*   x_feat     = (const int*)d_in[0];
  const int*   edge_index = (const int*)d_in[1];
  const int*   edge_attr  = (const int*)d_in[2];
  const float* atom_emb   = (const float*)d_in[3];
  const float* bond_emb   = (const float*)d_in[4];
  const float* eps        = (const float*)d_in[5];
  const float* W1         = (const float*)d_in[6];
  const float* b1         = (const float*)d_in[7];
  const float* g1         = (const float*)d_in[8];
  const float* be1        = (const float*)d_in[9];
  const float* W2         = (const float*)d_in[10];
  const float* b2         = (const float*)d_in[11];
  const float* g2         = (const float*)d_in[12];
  const float* be2        = (const float*)d_in[13];

  const int N = in_sizes[0] / 9;
  const int E = in_sizes[1] / 2;

  const int KP1 = 320, OUT1 = 600, NP1 = 608;  // GEMM1: K 300->320, cols 600->608
  const int KP2 = 608, OUT2 = 300, NP2 = 304;  // GEMM2: K 600->608, cols 300->304

  // workspace layout (all chunk sizes multiples of 16B)
  char* p = (char*)d_ws;
  float*  zf    = (float*)p;  p += (size_t)N * EMB  * 4;  // z, later aliased as T2
  __bf16* Zb    = (__bf16*)p; p += (size_t)N * KP1  * 2;
  float*  T1    = (float*)p;  p += (size_t)N * OUT1 * 4;
  __bf16* U     = (__bf16*)p; p += (size_t)N * KP2  * 2;
  __bf16* Wt1   = (__bf16*)p; p += (size_t)NP1 * KP1 * 2;
  __bf16* Wt2   = (__bf16*)p; p += (size_t)NP2 * KP2 * 2;
  float*  sums1 = (float*)p;  p += 2 * OUT1 * 4;
  float*  sums2 = (float*)p;  p += 2 * OUT2 * 4;

  float* h  = (float*)d_out;   // running node features; final layer lands here
  float* T2 = zf;              // reuse z buffer for second GEMM output

  const int mtiles   = (N + 15) / 16;
  const int ntilesN1 = NP1 / 16;                    // 38
  const int ntilesN2 = NP2 / 16;                    // 19
  const long long waves1 = (long long)mtiles * ntilesN1;
  const long long waves2 = (long long)mtiles * ntilesN2;
  const unsigned gblocks1 = (unsigned)((waves1 + 7) / 8);
  const unsigned gblocks2 = (unsigned)((waves2 + 7) / 8);
  const float invN = 1.f / (float)N;
  const int rowsPerBlk = 250;
  const unsigned statBlks = (unsigned)((N + rowsPerBlk - 1) / rowsPerBlk);

  // atom encoder -> h
  atom_enc_kernel<<<GRID1D((long long)N * EMB), 256, 0, stream>>>(
      x_feat, atom_emb, h, (long long)N * EMB);

  for (int l = 0; l < 5; ++l) {
    // z = (1+eps)*h, then scatter-add relu(h[src]+bond)
    zinit_kernel<<<GRID1D((long long)N * EMB), 256, 0, stream>>>(
        h, eps, l, zf, (long long)N * EMB);
    message_kernel<<<GRID1D((long long)E * EMB), 256, 0, stream>>>(
        edge_index, edge_attr, bond_emb + (size_t)l * 3 * 6 * EMB, h, zf,
        E, (long long)E * EMB);

    // bf16 convert + pad, prep weights, GEMM1
    cvt_pad_kernel<<<GRID1D((long long)N * KP1), 256, 0, stream>>>(
        zf, Zb, EMB, KP1, (long long)N * KP1);
    prep_w_kernel<<<GRID1D((long long)NP1 * KP1), 256, 0, stream>>>(
        W1 + (size_t)l * EMB * OUT1, Wt1, EMB, OUT1, KP1, (long long)NP1 * KP1);
    gemm_bf16_kernel<<<gblocks1, 256, 0, stream>>>(
        Zb, Wt1, b1 + (size_t)l * OUT1, T1, N, KP1, OUT1, ntilesN1);

    // BN1 stats + fused normalize/ReLU/convert -> U
    zero_kernel<<<GRID1D(2 * OUT1), 256, 0, stream>>>(sums1, 2 * OUT1);
    colstats_kernel<<<statBlks, 256, 0, stream>>>(T1, sums1, N, OUT1, rowsPerBlk);
    bn_relu_cvt_kernel<<<GRID1D((long long)N * KP2), 256, 0, stream>>>(
        T1, sums1, g1 + (size_t)l * OUT1, be1 + (size_t)l * OUT1, U,
        OUT1, KP2, invN, (long long)N * KP2);

    // GEMM2
    prep_w_kernel<<<GRID1D((long long)NP2 * KP2), 256, 0, stream>>>(
        W2 + (size_t)l * OUT1 * OUT2, Wt2, OUT1, OUT2, KP2, (long long)NP2 * KP2);
    gemm_bf16_kernel<<<gblocks2, 256, 0, stream>>>(
        U, Wt2, b2 + (size_t)l * OUT2, T2, N, KP2, OUT2, ntilesN2);

    // BN2 stats + normalize (+ReLU except last layer) -> h
    zero_kernel<<<GRID1D(2 * OUT2), 256, 0, stream>>>(sums2, 2 * OUT2);
    colstats_kernel<<<statBlks, 256, 0, stream>>>(T2, sums2, N, OUT2, rowsPerBlk);
    bn_final_kernel<<<GRID1D((long long)N * EMB), 256, 0, stream>>>(
        T2, sums2, g2 + (size_t)l * OUT2, be2 + (size_t)l * OUT2, h,
        OUT2, invN, (l < 4) ? 1 : 0, (long long)N * EMB);
  }
}
